// LFGA_9363028706078
// MI455X (gfx1250) — matrix operand order repaired and checked by reading, not compiled, not run on hardware
//
#include <hip/hip_runtime.h>

// ---------------------------------------------------------------------------
// B=8, C=64, H=W=64 -> N=4096, C_qk=16.
//   q=Wq*fb+bq, k=Wk*fb+bk, v=Wv*fa+bv
//   attn = softmax(q^T k) over keys; out = relu(gamma * (v attn^T) + fa)
// Fused flash-attention with bf16 WMMA (f32 accum), async-to-LDS staging.
// ---------------------------------------------------------------------------
#define BATCH 8
#define CHN   64
#define CQK   16
#define NTOK  4096
#define MTILE 32                 // keys per inner tile (full WMMA K-depth)
#define QBLK  64                 // queries per block (4 waves x 16)
#define NTILES (NTOK / MTILE)    // 128

typedef unsigned short u16;
typedef __attribute__((ext_vector_type(16))) __bf16        bf16x16;
typedef __attribute__((ext_vector_type(8)))  float         floatx8;
typedef __attribute__((ext_vector_type(4)))  unsigned int  uint4v;

union Frag {
    bf16x16 v;
    uint4v  q[2];
};

__device__ __forceinline__ u16 f2bf(float f) {
    unsigned int u = __float_as_uint(f);
    unsigned int r = u + 0x7FFFu + ((u >> 16) & 1u);   // round-to-nearest-even
    return (u16)(r >> 16);
}

// gfx1250 async copy: global -> LDS, 16 bytes per lane, tracked by ASYNCcnt.
__device__ __forceinline__ void async_copy16(unsigned lds_off, const void* gptr) {
    asm volatile("global_load_async_to_lds_b128 %0, %1, off"
                 :: "v"(lds_off), "v"(gptr) : "memory");
}
__device__ __forceinline__ void wait_async0() {
    asm volatile("s_wait_asynccnt 0x0" ::: "memory");
}

// ---------------------------------------------------------------------------
// Kernel 1: fused 1x1-conv projections, fp32 accumulate, bf16 store.
//   qws [B][N][16]  (token rows contiguous, 32B: S-matmul A operand)
//   kws [B][N][16]  (token rows contiguous, 32B: S-matmul B operand)
//   vws [B][64][N]  (channel-major: O-matmul A operand contiguous along m)
// ---------------------------------------------------------------------------
__global__ __launch_bounds__(256) void qkv_proj_kernel(
    const float* __restrict__ fa, const float* __restrict__ fb,
    const float* __restrict__ Wq, const float* __restrict__ bq,
    const float* __restrict__ Wk, const float* __restrict__ bk,
    const float* __restrict__ Wv, const float* __restrict__ bv,
    u16* __restrict__ qws, u16* __restrict__ kws, u16* __restrict__ vws)
{
    const int gid = blockIdx.x * 256 + threadIdx.x;     // one token per thread
    const int b = gid >> 12;
    const int n = gid & (NTOK - 1);

    float qa[CQK], ka[CQK], va[CHN];
#pragma unroll
    for (int o = 0; o < CQK; ++o) { qa[o] = bq[o]; ka[o] = bk[o]; }
#pragma unroll
    for (int o = 0; o < CHN; ++o) { va[o] = bv[o]; }

    const float* fbp = fb + (size_t)b * CHN * NTOK + n;
    const float* fap = fa + (size_t)b * CHN * NTOK + n;

    for (int c = 0; c < CHN; ++c) {
        const float xb = fbp[(size_t)c * NTOK];
        const float xa = fap[(size_t)c * NTOK];
#pragma unroll
        for (int o = 0; o < CQK; ++o) {                  // weights lane-invariant
            qa[o] = fmaf(Wq[o * CHN + c], xb, qa[o]);
            ka[o] = fmaf(Wk[o * CHN + c], xb, ka[o]);
        }
#pragma unroll
        for (int o = 0; o < CHN; ++o) {
            va[o] = fmaf(Wv[o * CHN + c], xa, va[o]);
        }
    }

    unsigned int* qrow = (unsigned int*)(qws + ((size_t)(b * NTOK + n) * CQK));
    unsigned int* krow = (unsigned int*)(kws + ((size_t)(b * NTOK + n) * CQK));
#pragma unroll
    for (int o = 0; o < CQK / 2; ++o) {
        qrow[o] = (unsigned int)f2bf(qa[2 * o]) | ((unsigned int)f2bf(qa[2 * o + 1]) << 16);
        krow[o] = (unsigned int)f2bf(ka[2 * o]) | ((unsigned int)f2bf(ka[2 * o + 1]) << 16);
    }
#pragma unroll
    for (int o = 0; o < CHN; ++o) {
        vws[(size_t)(b * CHN + o) * NTOK + n] = f2bf(va[o]);
    }
}

// ---------------------------------------------------------------------------
// Kernel 2: fused flash attention + epilogue.
//   128 threads (4 wave32), 16 queries/wave, 32-key inner tiles.
//   Per tile: 2x S-wmma (K=16ch padded) + 4x O-wmma (full K=32 keys).
//   K/V tiles double-buffered in LDS via global_load_async_to_lds_b128.
// ---------------------------------------------------------------------------
__global__ __launch_bounds__(128) void attn_kernel(
    const u16* __restrict__ qws, const u16* __restrict__ kws,
    const u16* __restrict__ vws, const float* __restrict__ fa,
    const float* __restrict__ gamma, float* __restrict__ out)
{
    const int tid  = threadIdx.x;
    const int wave = tid >> 5;
    const int lane = tid & 31;
    const int l16  = lane & 15;
    const int hl   = lane >> 4;              // 0: lanes 0-15, 1: lanes 16-31

    const int nBlk = NTOK / QBLK;
    const int b    = blockIdx.x / nBlk;
    const int n0   = (blockIdx.x % nBlk) * QBLK;
    const int nq   = n0 + wave * 16;

    __shared__ __attribute__((aligned(16))) u16  kTile[2][MTILE * CQK];   // 2x1KB
    __shared__ __attribute__((aligned(16))) u16  vTile[2][CHN * MTILE];   // 2x4KB
    __shared__ __attribute__((aligned(16))) u16  pTile[4][16 * MTILE];    // 4x1KB
    __shared__ float statLds[4][16];

    // LDS byte offsets (low 32 bits of the generic shared pointer)
    const unsigned kbase = (unsigned)(size_t)&kTile[0][0];
    const unsigned vbase = (unsigned)(size_t)&vTile[0][0];

    const float gm = *gamma;
    const uint4v z4 = (uint4v)0u;

    // Q A-fragment (K=16 channels, regs 4-7 zero-padded), loaded once.
    Frag aq;
    {
        const u16* src = qws + ((size_t)(b * NTOK + nq + l16) * CQK) + (hl ? 8 : 0);
        aq.q[0] = *(const uint4v*)src;
        aq.q[1] = z4;
    }

    floatx8 oacc[4];
#pragma unroll
    for (int ct = 0; ct < 4; ++ct) oacc[ct] = (floatx8)0.0f;

    float runMax[8], runSum[8];
#pragma unroll
    for (int r = 0; r < 8; ++r) { runMax[r] = -1.0e30f; runSum[r] = 0.0f; }

    // ---- async stage of tile (bf, m0): K 1KB + V 4KB, 16B chunks ----
    auto stage = [&](int bf, int m0) {
        if (tid < 64) {   // kTile: 64 chunks; row = chunk/2, half-row = chunk&1
            const unsigned lds = kbase + (unsigned)bf * (MTILE * CQK * 2) + tid * 16;
            const u16* g = kws + ((size_t)(b * NTOK + m0 + (tid >> 1)) * CQK) + (tid & 1) * 8;
            async_copy16(lds, g);
        }
#pragma unroll
        for (int i = 0; i < 2; ++i) {   // vTile: 256 chunks; row = chunk/4
            const int c = tid + i * 128;
            const unsigned lds = vbase + (unsigned)bf * (CHN * MTILE * 2) + c * 16;
            const u16* g = vws + ((size_t)(b * CHN + (c >> 2)) * NTOK) + m0 + (c & 3) * 8;
            async_copy16(lds, g);
        }
    };

    stage(0, 0);   // prime the pipeline

    for (int t = 0; t < NTILES; ++t) {
        const int bf = t & 1;
        wait_async0();                       // this wave's staged data in LDS
        __syncthreads();                     // everyone's staged data visible
        if (t + 1 < NTILES) stage(bf ^ 1, (t + 1) * MTILE);

        // ---- S = Q^T K: two 16-key column tiles ----
        const uint4v* kt = (const uint4v*)kTile[bf];     // 2 x uint4v per token
        Frag bk0, bk1;
        bk0.q[0] = kt[l16 * 2 + hl];
        bk0.q[1] = z4;
        bk1.q[0] = kt[(16 + l16) * 2 + hl];
        bk1.q[1] = z4;

        floatx8 zc = (floatx8)0.0f;
        floatx8 s0 = __builtin_amdgcn_wmma_f32_16x16x32_bf16(
            false, aq.v, false, bk0.v, (short)0, zc, false, false);
        floatx8 s1 = __builtin_amdgcn_wmma_f32_16x16x32_bf16(
            false, aq.v, false, bk1.v, (short)0, zc, false, false);

        // ---- online softmax across 32 key columns ----
        float sc[8];
#pragma unroll
        for (int r = 0; r < 8; ++r) {
            float tm = fmaxf(s0[r], s1[r]);
            tm = fmaxf(tm, __shfl_xor(tm, 1));
            tm = fmaxf(tm, __shfl_xor(tm, 2));
            tm = fmaxf(tm, __shfl_xor(tm, 4));
            tm = fmaxf(tm, __shfl_xor(tm, 8));
            const float nm = fmaxf(runMax[r], tm);
            const float f  = __expf(runMax[r] - nm);
            const float p0 = __expf(s0[r] - nm);
            const float p1 = __expf(s1[r] - nm);
            float rs = p0 + p1;
            rs += __shfl_xor(rs, 1);
            rs += __shfl_xor(rs, 2);
            rs += __shfl_xor(rs, 4);
            rs += __shfl_xor(rs, 8);
            runSum[r] = runSum[r] * f + rs;
            runMax[r] = nm;
            sc[r]     = f;
            // transpose P into LDS: row n = r + 8*hl, cols m = l16, 16+l16
            const int rowoff = (r + hl * 8) * MTILE;
            pTile[wave][rowoff + l16]      = f2bf(p0);
            pTile[wave][rowoff + 16 + l16] = f2bf(p1);
        }
        if (l16 == 0) {
#pragma unroll
            for (int r = 0; r < 8; ++r) statLds[wave][r + hl * 8] = sc[r];
        }
        __syncthreads();

        const float fcol = statLds[wave][l16];   // rescale for column n = l16

        // P^T B-fragment, full K=32: lane reads row n=l16, keys hl*8..+8 and 16+hl*8..+8
        Frag bp;
        {
            const uint4v* pt = (const uint4v*)pTile[wave];
            bp.q[0] = pt[l16 * 4 + hl];
            bp.q[1] = pt[l16 * 4 + hl + 2];
        }

        // ---- O = O*fcol + V P^T: 4 channel tiles, full K=32 ----
#pragma unroll
        for (int ct = 0; ct < 4; ++ct) {
            Frag av;   // A 16x32: regs0-3 = m (hl?8:0)..+8, regs4-7 = +16
            const u16* vr = vTile[bf] + (ct * 16 + l16) * MTILE + (hl ? 8 : 0);
            av.q[0] = *(const uint4v*)vr;
            av.q[1] = *(const uint4v*)(vr + 16);
            floatx8 c = oacc[ct];
#pragma unroll
            for (int r = 0; r < 8; ++r) c[r] *= fcol;
            oacc[ct] = __builtin_amdgcn_wmma_f32_16x16x32_bf16(
                false, av.v, false, bp.v, (short)0, c, false, false);
        }
    }

    // ---- normalize, gamma, residual, relu ----
    __syncthreads();
    if (l16 == 0) {
#pragma unroll
        for (int r = 0; r < 8; ++r) statLds[wave][r + hl * 8] = runSum[r];
    }
    __syncthreads();
    const float inv = gm / statLds[wave][l16];

    const int ncol = nq + l16;
#pragma unroll
    for (int ct = 0; ct < 4; ++ct) {
#pragma unroll
        for (int r = 0; r < 8; ++r) {
            const int ch = ct * 16 + r + hl * 8;
            const size_t idx = (size_t)(b * CHN + ch) * NTOK + ncol;
            const float val = oacc[ct][r] * inv + fa[idx];
            out[idx] = fmaxf(val, 0.0f);
        }
    }
}

// ---------------------------------------------------------------------------
// Host launcher
// ---------------------------------------------------------------------------
extern "C" void kernel_launch(void* const* d_in, const int* in_sizes, int n_in,
                              void* d_out, int out_size, void* d_ws, size_t ws_size,
                              hipStream_t stream) {
    (void)in_sizes; (void)n_in; (void)out_size; (void)ws_size;

    const float* fa    = (const float*)d_in[0];
    const float* fb    = (const float*)d_in[1];
    const float* Wq    = (const float*)d_in[2];
    const float* bq    = (const float*)d_in[3];
    const float* Wk    = (const float*)d_in[4];
    const float* bk    = (const float*)d_in[5];
    const float* Wv    = (const float*)d_in[6];
    const float* bv    = (const float*)d_in[7];
    const float* gamma = (const float*)d_in[8];
    float*       out   = (float*)d_out;

    // workspace: q 1MB, k 1MB, v 4MB (bf16)
    u16* qws = (u16*)d_ws;
    u16* kws = qws + (size_t)BATCH * NTOK * CQK;
    u16* vws = kws + (size_t)BATCH * NTOK * CQK;

    qkv_proj_kernel<<<(BATCH * NTOK) / 256, 256, 0, stream>>>(
        fa, fb, Wq, bq, Wk, bk, Wv, bv, qws, kws, vws);

    attn_kernel<<<BATCH * (NTOK / QBLK), 128, 0, stream>>>(
        qws, kws, vws, fa, gamma, out);
}
